// S4BaseBlock_84937273245931
// MI455X (gfx1250) — matrix operand order repaired and checked by reading, not compiled, not run on hardware
//
#include <hip/hip_runtime.h>

typedef __attribute__((ext_vector_type(16))) _Float16 v16h;
typedef __attribute__((ext_vector_type(8)))  _Float16 v8h;
typedef __attribute__((ext_vector_type(8)))  float    v8f;

#define S4_DT 0.1f
constexpr int NST  = 64;    // state dim
constexpr int DCH  = 128;   // channels
constexpr int BSZ  = 16;    // batch
constexpr int LSEQ = 4096;  // sequence length
constexpr int TAPS = 64;    // FIR taps: rho(Abar) <= 0.56 -> 0.56^64 ~ 8e-17 < f32 eps

constexpr int TSTRIP = 256;            // output timesteps per workgroup
constexpr int WIN    = TSTRIP + 48;    // staged window (48-tap halo, rest in-tile)
constexpr int WPAD   = WIN + 8;        // padded LDS row stride (halves)

// xor-butterfly lane swap via ds_swizzle (imm must be a literal)
#define SWZF(v, imm) __builtin_bit_cast(float, \
    __builtin_amdgcn_ds_swizzle(__builtin_bit_cast(int, (v)), (imm)))

// ---------------------------------------------------------------------------
// Kernel 0: discretize + FIR taps. One block (128 thr) per channel d.
// 1) Gauss-Jordan on [I-dt/2*A | I+dt/2*A | dt*B]  (diag dominant, no pivot)
//    -> Abar in cols 64..127, Bbar in col 128.
// 2) kappa_d[delta] = C . Abar^delta . Bbar for delta = 0..63 by iterated
//    matvec; each thread keeps its Abar row in registers.
// ---------------------------------------------------------------------------
__global__ void __launch_bounds__(128) s4_taps(
    const float* __restrict__ A,      // [D][64][64]
    const float* __restrict__ Bin,    // [D][64][1]
    const float* __restrict__ Cin,    // [D][1][64]
    float* __restrict__ kapOut)       // ws: [D][64]
{
  __shared__ float aug[NST][132];
  __shared__ float colk[NST];
  __shared__ float vvec[NST];
  __shared__ float part[2];
  const int d   = blockIdx.x;
  const int tid = threadIdx.x;
  const float* Ad = A + (size_t)d * NST * NST;

  for (int idx = tid; idx < NST * 132; idx += 128) {
    int i = idx / 132, j = idx % 132;
    float v = 0.0f;
    if (j < NST)            v = (i == j ? 1.0f : 0.0f) - 0.5f * S4_DT * Ad[i * NST + j];
    else if (j < 2 * NST)   v = ((j - NST) == i ? 1.0f : 0.0f) + 0.5f * S4_DT * Ad[i * NST + (j - NST)];
    else if (j == 2 * NST)  v = S4_DT * Bin[d * NST + i];
    aug[i][j] = v;
  }
  __syncthreads();

  for (int k = 0; k < NST; ++k) {
    float pivinv = 1.0f / aug[k][k];
    float rk0 = aug[k][tid] * pivinv;
    float rk1 = (tid == 0) ? aug[k][128] * pivinv : 0.0f;
    if (tid < NST) colk[tid] = aug[tid][k];
    __syncthreads();
    aug[k][tid] = rk0;
    if (tid == 0) aug[k][128] = rk1;
    for (int i = 0; i < NST; ++i) {
      if (i == k) continue;
      float m = colk[i];
      aug[i][tid] -= m * rk0;
      if (tid == 0) aug[i][128] -= m * rk1;
    }
    __syncthreads();
  }

  // --- tap recursion: v <- Abar v ; kappa = C.v ---
  float arow[NST];
  float cv_i = 0.0f;
  if (tid < NST) {
#pragma unroll
    for (int t = 0; t < NST; ++t) arow[t] = aug[tid][NST + t];  // my Abar row
    vvec[tid] = aug[tid][128];                                  // v0 = Bbar
    cv_i = Cin[d * NST + tid];
  }
  __syncthreads();

  for (int del = 0; del < TAPS; ++del) {
    float p = 0.0f, s = 0.0f;
    if (tid < NST) {
      p = cv_i * vvec[tid];
#pragma unroll
      for (int t = 0; t < NST; ++t) s += arow[t] * vvec[t];
    }
    // reduce p across the two active waves
    p += SWZF(p, 0x041F);  // xor 1
    p += SWZF(p, 0x081F);  // xor 2
    p += SWZF(p, 0x101F);  // xor 4
    p += SWZF(p, 0x201F);  // xor 8
    p += SWZF(p, 0x401F);  // xor 16
    if (tid == 0)  part[0] = p;
    if (tid == 32) part[1] = p;
    __syncthreads();                 // vvec reads done; parts visible
    if (tid < NST) vvec[tid] = s;
    if (tid == 0) kapOut[d * TAPS + del] = part[0] + part[1];
    __syncthreads();
  }
}

// ---------------------------------------------------------------------------
// Kernel 1: banded-Toeplitz FIR via WMMA. Grid = (D, LSEQ/TSTRIP); 4 wave32.
// y[b,t,d] = sum_{delta=0..63} kappa[d][delta] * x[b,t-delta,d]
// Per 16-step tile: Y(16x16) = T(16x64) @ X(64x16), T[i][k] = kappa[i+48-k],
// done as 2x v_wmma_f32_16x16x32_f16 from an LDS-staged f16 x window.
// ---------------------------------------------------------------------------
__global__ void __launch_bounds__(128) s4_fir(
    const float* __restrict__ x,     // [16][4096][128]
    const float* __restrict__ kapWs, // ws [128][64]
    float* __restrict__ y)           // [16][4096][128]
{
  const int d     = blockIdx.x;
  const int strip = blockIdx.y;
  const int base  = strip * TSTRIP;
  const int tid   = threadIdx.x;
  const int w     = tid >> 5;
  const int lane  = tid & 31;
  const int n     = lane & 15;       // batch column / Toeplitz row
  const int lh    = lane >> 4;

  __shared__ _Float16 X[BSZ][WPAD] __attribute__((aligned(16)));
  __shared__ float kap[TAPS];

  if (tid < TAPS) kap[tid] = kapWs[d * TAPS + tid];

  // --- stage x window [base-48, base+TSTRIP) x 16 batches as f16 ---
  for (int idx = tid; idx < BSZ * WIN; idx += 128) {
    int b = idx / WIN, tt = idx % WIN;
    int t = base - 48 + tt;
    float v = (t >= 0) ? x[(size_t)b * (LSEQ * DCH) + (size_t)t * DCH + d] : 0.0f;
    X[b][tt] = (_Float16)v;
  }
  __syncthreads();

  // --- build Toeplitz A operands from taps (identical for all waves) ---
  // A-operand k order used here (contiguous, same as B; not the ISA table order): lane m(0-15): K=0..15; lanes 16-31: K=16..31.
  v16h T0, T1;
#pragma unroll
  for (int e = 0; e < 16; ++e) {
    int kp = e + (lh << 4);   // K within 32-chunk
    int i0 = n + 48 - kp;        // tap index, chunk 0 (k = kp):      17..63
    int i1 = i0 - 32;            // tap index, chunk 1 (k = 32 + kp): -15..31
    T0[e] = (_Float16)kap[i0];
    T1[e] = (i1 >= 0) ? (_Float16)kap[i1] : (_Float16)0.0f;
  }

  // --- 4 independent tiles per wave ---
#pragma unroll
  for (int it = 0; it < 4; ++it) {
    const int tile = 4 * w + it;             // 16 tiles per WG
    const int t0   = base + 16 * tile;
    const int ws0  = 16 * tile;              // window offset of k=0 (= t0-48 in lds coords)

    // B operand k order used here (contiguous, same as A; not the ISA table order): lane n(0-15): K=0..15; lanes 16-31: K=16..31.
    v8h p0 = *(const v8h*)&X[n][ws0 + 16 * lh];
    v8h p1 = *(const v8h*)&X[n][ws0 + 16 * lh + 8];
    v8h p2 = *(const v8h*)&X[n][ws0 + 32 + 16 * lh];
    v8h p3 = *(const v8h*)&X[n][ws0 + 32 + 16 * lh + 8];
    v16h b0, b1;
#pragma unroll
    for (int e = 0; e < 8; ++e) {
      b0[e] = p0[e]; b0[e + 8] = p1[e];
      b1[e] = p2[e]; b1[e + 8] = p3[e];
    }

    v8f c = {};
    c = __builtin_amdgcn_wmma_f32_16x16x32_f16(false, T0, false, b0, (short)0, c, false, false);
    c = __builtin_amdgcn_wmma_f32_16x16x32_f16(false, T1, false, b1, (short)0, c, false, false);

    // C/D layout: lane col = n (batch), rows = r + 8*lh (step within tile)
#pragma unroll
    for (int r = 0; r < 8; ++r) {
      y[(size_t)n * (LSEQ * DCH) + (size_t)(t0 + 8 * lh + r) * DCH + d] = c[r];
    }
  }
}

// ---------------------------------------------------------------------------
extern "C" void kernel_launch(void* const* d_in, const int* in_sizes, int n_in,
                              void* d_out, int out_size, void* d_ws, size_t ws_size,
                              hipStream_t stream) {
  (void)in_sizes; (void)n_in; (void)out_size; (void)ws_size;
  const float* x  = (const float*)d_in[0];
  const float* A  = (const float*)d_in[1];
  const float* Bm = (const float*)d_in[2];
  const float* Cm = (const float*)d_in[3];
  float* y = (float*)d_out;

  float* kapWs = (float*)d_ws;   // 128*64 f32 = 32 KB

  s4_taps<<<dim3(DCH), dim3(128), 0, stream>>>(A, Bm, Cm, kapWs);
  s4_fir<<<dim3(DCH, LSEQ / TSTRIP), dim3(128), 0, stream>>>(x, kapWs, y);
}
